// MultiHeadSelfAttentionTENER_4071628997264
// MI455X (gfx1250) — matrix operand order repaired and checked by reading, not compiled, not run on hardware
//
#include <hip/hip_runtime.h>
#include <hip/hip_fp16.h>

// ---------------------------------------------------------------------------
// TENER multi-head self-attention, fused flash-style, CDNA5 (gfx1250) WMMA.
// B=4, T=1024, D=512, H=8, dh=vdh=64.
// shift(t)[q,k] = t[q, T+k-q]  =>  attn[q,k] = q.k + q.pe[T+k-q] + vb[h,T+k-q]
// ---------------------------------------------------------------------------

typedef _Float16 f16;
typedef __attribute__((ext_vector_type(16))) _Float16 v16h;
typedef __attribute__((ext_vector_type(8)))  _Float16 v8h;
typedef __attribute__((ext_vector_type(8)))  float    v8f;

#define DEV __device__ __forceinline__

static constexpr int T    = 1024;
static constexpr int TT   = 2048;   // 2T
static constexpr int TTP  = 2064;   // 2T + 16 zero pad rows (kills clamps in hot loop)
static constexpr int D    = 512;
static constexpr int H    = 8;
static constexpr int DH   = 64;     // per-head dim
static constexpr int Bsz  = 4;

DEV v8f wmma_f16(v16h a, v16h b, v8f c) {
  return __builtin_amdgcn_wmma_f32_16x16x32_f16(
      /*neg_a=*/false, a, /*neg_b=*/false, b,
      /*c_mod=*/(short)0, c, /*reuse_a=*/false, /*reuse_b=*/false);
}

// Fragment built from two 16B loads that land directly in the v16h registers.
union Frag { v16h v; v8h h[2]; };

// A-fragment (16x32 f16): lane<16 -> M=lane, K in {0..7, 16..23}; lane>=16 -> K {8..15, 24..31}
template<typename P>
DEV v16h lda_frag(P base, int ld, int ks) {
  int lane = threadIdx.x & 31;
  auto p = base + (lane & 15) * ld + ks + ((lane >> 4) << 3);
  Frag f;
  f.h[0] = *(const v8h*)p;
  f.h[1] = *(const v8h*)(p + 16);
  return f.v;
}

// B-fragment (32x16 f16): lane%16 = column N; lane<16 -> K {0..7, 16..23}; lane>=16 -> K {8..15, 24..31}.
// Column N data must be contiguous at base + N*ld (i.e. B stored transposed/row-per-column).
template<typename P>
DEV v16h ldb_frag(P base, int ld, int ks) {
  int lane = threadIdx.x & 31;
  auto p = base + (size_t)(lane & 15) * ld + ks + ((lane >> 4) << 3);
  Frag f;
  f.h[0] = *(const v8h*)p;
  f.h[1] = *(const v8h*)(p + 16);
  return f.v;
}

DEV float rmax16(float v) {
#pragma unroll
  for (int m = 1; m < 16; m <<= 1) v = fmaxf(v, __shfl_xor(v, m, 32));
  return v;
}
DEV float rsum16(float v) {
#pragma unroll
  for (int m = 1; m < 16; m <<= 1) v += __shfl_xor(v, m, 32);
  return v;
}

// ---------------------------------------------------------------------------
// Prep kernels (negligible runtime; kept compact)
// ---------------------------------------------------------------------------

// pe[l][d] (f16) for l in [0,2T) (+16 zero rows); vbr[h][l] = v_bias[h] . pe[l]
__global__ __launch_bounds__(256) void pe_vbias_kernel(const float* __restrict__ v_bias,
                                                       f16* __restrict__ pe,
                                                       float* __restrict__ vbr) {
  int l = blockIdx.x * blockDim.x + threadIdx.x;
  if (l >= TTP) return;
  if (l >= TT) {            // zero pad rows
#pragma unroll 4
    for (int d = 0; d < DH; ++d) pe[(size_t)l * DH + d] = (f16)0.f;
    return;
  }
  const float fac = -9.210340371976184f / 31.0f;   // -ln(10000)/(half-1), half=32
  float pos = (float)l - (float)T;
  float row[DH];
#pragma unroll 4
  for (int j = 0; j < 32; ++j) {
    float ang = pos * expf((float)j * fac);
    row[j]      = sinf(ang);
    row[j + 32] = cosf(ang);
  }
#pragma unroll 4
  for (int d = 0; d < DH; ++d) pe[(size_t)l * DH + d] = (f16)row[d];
  for (int h = 0; h < H; ++h) {
    float s = 0.f;
#pragma unroll 4
    for (int d = 0; d < DH; ++d) s += v_bias[h * DH + d] * row[d];
    vbr[h * TT + l] = s;
  }
}

// WT[n][k] = (f16) W[k][n]   (512x512)
__global__ __launch_bounds__(256) void transpose_w_kernel(const float* __restrict__ W,
                                                          f16* __restrict__ WT) {
  int idx = blockIdx.x * blockDim.x + threadIdx.x;   // n*512 + k
  int n = idx >> 9, k = idx & 511;
  WT[idx] = (f16)W[k * D + n];
}

// kf[b][h][t][d] = (f16) key[b][t][h*64+d]
__global__ __launch_bounds__(256) void convert_k_kernel(const float* __restrict__ key,
                                                        f16* __restrict__ kf) {
  size_t i = (size_t)blockIdx.x * blockDim.x + threadIdx.x;   // over B*T*D
  int c = (int)(i % D);
  int t = (int)((i / D) % T);
  int b = (int)(i / ((size_t)D * T));
  int h = c >> 6, d = c & 63;
  kf[((((size_t)b * H + h) * T + t) << 6) + d] = (f16)key[i];
}

// ---------------------------------------------------------------------------
// WMMA GEMM: C[M=4096][N=512] = A[4096][512] @ B[512][512] (+ bias)
// BT is f16, row n holds column n of B (contiguous K).
// MODE 0: A fp32, out f16 -> q layout [b][h][t][dh]
// MODE 1: A fp32, out f16 -> vT layout [b][h][dh][t]
// MODE 2: A f16,  out fp32 -> d_out [row][col] (+bias)
// Block: 256 thr = 8 waves; wave w does rows [blk*128 + w*16, +16), cols [blkY*64, +64).
// ---------------------------------------------------------------------------
template<int MODE>
__global__ __launch_bounds__(256) void gemm_wmma_kernel(const float* __restrict__ Af,
                                                        const f16* __restrict__ Ah,
                                                        const f16* __restrict__ BT,
                                                        const float* __restrict__ bias,
                                                        f16* __restrict__ outH,
                                                        float* __restrict__ outF) {
  int lane = threadIdx.x & 31, wave = threadIdx.x >> 5;
  int row0 = blockIdx.x * 128 + wave * 16;
  int n0   = blockIdx.y * 64;
  v8f acc[4];
#pragma unroll
  for (int j = 0; j < 4; ++j)
#pragma unroll
    for (int r = 0; r < 8; ++r) acc[j][r] = 0.f;

  int m = lane & 15;
  int kb = (lane >> 4) << 3;
  for (int k0 = 0; k0 < D; k0 += 32) {
    v16h a;
    if (MODE == 2) {
      a = lda_frag(Ah + (size_t)row0 * D, D, k0);
    } else {
      const float* p = Af + (size_t)(row0 + m) * D + k0 + kb;
#pragma unroll
      for (int i = 0; i < 8; ++i) { a[i] = (f16)p[i]; a[i + 8] = (f16)p[i + 16]; }
    }
#pragma unroll
    for (int j = 0; j < 4; ++j) {
      v16h b = ldb_frag(BT + (size_t)(n0 + j * 16) * D, D, k0);
      acc[j] = wmma_f16(a, b, acc[j]);
    }
  }

  int nl = lane & 15, mbase = (lane >> 4) << 3;
#pragma unroll
  for (int j = 0; j < 4; ++j) {
    int col = n0 + j * 16 + nl;
    float bv = bias[col];
#pragma unroll
    for (int r = 0; r < 8; ++r) {
      int row = row0 + mbase + r;
      float val = acc[j][r] + bv;
      if (MODE == 0) {
        int b = row >> 10, t = row & 1023, h = col >> 6, d = col & 63;
        outH[((((size_t)b * H + h) * T + t) << 6) + d] = (f16)val;
      } else if (MODE == 1) {
        int b = row >> 10, t = row & 1023, h = col >> 6, d = col & 63;
        outH[((((size_t)b * H + h) * DH + d) << 10) + t] = (f16)val;
      } else {
        outF[(size_t)row * D + col] = val;
      }
    }
  }
}

// ---------------------------------------------------------------------------
// Fused TENER attention. grid = (T/128, B*H), block = 256 (8 waves, 16 q each).
// x[b][t][h*64+j] = softmax_k(q.k + q.pe[T+k-q] + vb[T+k-q]) @ v   (f16 out)
// ---------------------------------------------------------------------------
__global__ __launch_bounds__(256) void attn_tener_kernel(const f16* __restrict__ qf,
                                                         const f16* __restrict__ kf,
                                                         const f16* __restrict__ vT,
                                                         const f16* __restrict__ pe,
                                                         const float* __restrict__ vbr,
                                                         f16* __restrict__ xout) {
  __shared__ float vb[TTP];                      // 8.25 KB: vbias_rel row (+ zero pad)
  __shared__ float Rt[8][16 * 48];               // 24 KB: per-wave rel window (16x48)
  __shared__ __align__(16) f16 Pt[8][16 * 32];   // 8 KB: per-wave P tile for A-frag reload

  int lane = threadIdx.x & 31, w = threadIdx.x >> 5;
  int bh = blockIdx.y;
  int b = bh >> 3, h = bh & 7;
  int q0 = blockIdx.x * 128 + w * 16;
  int nl = lane & 15, mbase = (lane >> 4) << 3;

  for (int i = threadIdx.x; i < TTP; i += 256) vb[i] = (i < TT) ? vbr[h * TT + i] : 0.f;
  __syncthreads();

  // Q fragments for this wave's 16 rows (dh=64 -> two K slices), loaded once.
  const f16* qbase = qf + (((size_t)bh * T + q0) << 6);
  v16h aq0 = lda_frag(qbase, DH, 0);
  v16h aq1 = lda_frag(qbase, DH, 32);

  float mr[8], lr[8];
  v8f O[4];
#pragma unroll
  for (int r = 0; r < 8; ++r) { mr[r] = -1e30f; lr[r] = 0.f; }
#pragma unroll
  for (int j = 0; j < 4; ++j)
#pragma unroll
    for (int r = 0; r < 8; ++r) O[j][r] = 0.f;

  for (int k0 = 0; k0 < T; k0 += 32) {
    // ---- scores S = q @ k^T for 16x32 tile -------------------------------
    v8f S0, S1;
#pragma unroll
    for (int r = 0; r < 8; ++r) { S0[r] = 0.f; S1[r] = 0.f; }
    const f16* kb0 = kf + (((size_t)bh * T + k0) << 6);
    S0 = wmma_f16(aq0, ldb_frag(kb0, DH, 0),  S0);
    S0 = wmma_f16(aq1, ldb_frag(kb0, DH, 32), S0);
    const f16* kb1 = kb0 + (16 << 6);
    S1 = wmma_f16(aq0, ldb_frag(kb1, DH, 0),  S1);
    S1 = wmma_f16(aq1, ldb_frag(kb1, DH, 32), S1);

    // ---- relative window R[m][i] = q_m . pe[l0+i], i in [0,48) -----------
    // vbias_rel[l0+i] is row-independent: folded in at the LDS store.
    int l0 = T + k0 - q0 - 15;     // >= 1; l0+47 <= 2T (pad rows make it safe)
#pragma unroll
    for (int t = 0; t < 3; ++t) {
      float seed = vb[l0 + t * 16 + nl];
      v8f Rc;
#pragma unroll
      for (int r = 0; r < 8; ++r) Rc[r] = 0.f;
      const f16* pb = pe + (size_t)(l0 + t * 16) * DH;
      Rc = wmma_f16(aq0, ldb_frag(pb, DH, 0),  Rc);
      Rc = wmma_f16(aq1, ldb_frag(pb, DH, 32), Rc);
#pragma unroll
      for (int r = 0; r < 8; ++r)
        Rt[w][(mbase + r) * 48 + t * 16 + nl] = Rc[r] + seed;
    }
    __syncthreads();

    // attn = S + R'[m][n - m + 15]  (index is k0/q0-independent)
#pragma unroll
    for (int r = 0; r < 8; ++r) {
      int m = mbase + r;
      S0[r] += Rt[w][m * 47 + nl + 15];
      S1[r] += Rt[w][m * 47 + nl + 31];
    }

    // ---- online softmax (rows live across 16-lane groups) ----------------
    float sc[8];
#pragma unroll
    for (int r = 0; r < 8; ++r) {
      float v = rmax16(fmaxf(S0[r], S1[r]));
      float mn = fmaxf(mr[r], v);
      sc[r] = __expf(mr[r] - mn);
      mr[r] = mn;
    }
#pragma unroll
    for (int r = 0; r < 8; ++r) {
      float p0 = __expf(S0[r] - mr[r]);
      float p1 = __expf(S1[r] - mr[r]);
      S0[r] = p0; S1[r] = p1;
      lr[r] = lr[r] * sc[r] + rsum16(p0 + p1);
    }
#pragma unroll
    for (int j = 0; j < 4; ++j)
#pragma unroll
      for (int r = 0; r < 8; ++r) O[j][r] *= sc[r];

    // ---- C-layout -> A-layout via LDS, then O += P @ V -------------------
#pragma unroll
    for (int r = 0; r < 8; ++r) {
      Pt[w][(mbase + r) * 32 + nl]      = (f16)S0[r];
      Pt[w][(mbase + r) * 32 + 16 + nl] = (f16)S1[r];
    }
    __syncthreads();
    v16h ap = lda_frag((const f16*)&Pt[w][0], 32, 0);
#pragma unroll
    for (int j = 0; j < 4; ++j) {
      const f16* vb0 = vT + ((((size_t)bh * DH + j * 16)) << 10) + k0;
      O[j] = wmma_f16(ap, ldb_frag(vb0, T, 0), O[j]);
    }
    __syncthreads();
  }

  // ---- epilogue: x[b][q][h*64 + c] = O / l ------------------------------
#pragma unroll
  for (int r = 0; r < 8; ++r) lr[r] = 1.0f / lr[r];
#pragma unroll
  for (int j = 0; j < 4; ++j) {
#pragma unroll
    for (int r = 0; r < 8; ++r) {
      int row = q0 + mbase + r;
      int col = h * DH + j * 16 + nl;
      xout[(((size_t)b * T + row) << 9) + col] = (f16)(O[j][r] * lr[r]);
    }
  }
}

// ---------------------------------------------------------------------------
// Host launch
// ---------------------------------------------------------------------------
extern "C" void kernel_launch(void* const* d_in, const int* in_sizes, int n_in,
                              void* d_out, int out_size, void* d_ws, size_t ws_size,
                              hipStream_t stream) {
  const float* query  = (const float*)d_in[0];
  const float* key    = (const float*)d_in[1];
  const float* value  = (const float*)d_in[2];
  const float* Wq     = (const float*)d_in[4];
  const float* bq     = (const float*)d_in[5];
  const float* Wv     = (const float*)d_in[6];
  const float* bv     = (const float*)d_in[7];
  const float* Wo     = (const float*)d_in[8];
  const float* bo     = (const float*)d_in[9];
  const float* v_bias = (const float*)d_in[10];
  float* out = (float*)d_out;

  char* ws = (char*)d_ws;
  const size_t MB = 1u << 20, KB = 1u << 10;
  f16*   qf  = (f16*)(ws);                              // 4 MiB  [B][H][T][64]
  f16*   kf  = (f16*)(ws + 4 * MB);                     // 4 MiB  [B][H][T][64]
  f16*   vTp = (f16*)(ws + 8 * MB);                     // 4 MiB  [B][H][64][T]
  f16*   x   = (f16*)(ws + 12 * MB);                    // 4 MiB  [B*T][512]
  f16*   pe  = (f16*)(ws + 16 * MB);                    // 258 KiB [2T+16][64]
  float* vbr = (float*)(ws + 16 * MB + 272 * KB);       // 64 KiB  [H][2T]
  f16*   WqT = (f16*)(ws + 16 * MB + 336 * KB);         // 512 KiB
  f16*   WvT = (f16*)(ws + 16 * MB + 848 * KB);         // 512 KiB
  f16*   WoT = (f16*)(ws + 16 * MB + 1360 * KB);        // 512 KiB

  pe_vbias_kernel<<<(TTP + 255) / 256, 256, 0, stream>>>(v_bias, pe, vbr);
  transpose_w_kernel<<<(D * D) / 256, 256, 0, stream>>>(Wq, WqT);
  transpose_w_kernel<<<(D * D) / 256, 256, 0, stream>>>(Wv, WvT);
  transpose_w_kernel<<<(D * D) / 256, 256, 0, stream>>>(Wo, WoT);
  convert_k_kernel<<<(Bsz * T * D) / 256, 256, 0, stream>>>(key, kf);

  dim3 ggrid(Bsz * T / 128, D / 64);
  gemm_wmma_kernel<0><<<ggrid, 256, 0, stream>>>(query, nullptr, WqT, bq, qf,  nullptr);
  gemm_wmma_kernel<1><<<ggrid, 256, 0, stream>>>(value, nullptr, WvT, bv, vTp, nullptr);

  attn_tener_kernel<<<dim3(T / 128, Bsz * H), 256, 0, stream>>>(qf, kf, vTp, pe, vbr, x);

  gemm_wmma_kernel<2><<<ggrid, 256, 0, stream>>>(nullptr, x, WoT, bo, nullptr, out);
}
